// SimGCN_66632122630369
// MI455X (gfx1250) — matrix operand order repaired and checked by reading, not compiled, not run on hardware
//
#include <hip/hip_runtime.h>
#include <hip/hip_bf16.h>

// ---------------------------------------------------------------------------
// Types for WMMA / TDM (CDNA5 / gfx1250, wave32)
// ---------------------------------------------------------------------------
typedef __attribute__((ext_vector_type(16))) __bf16          v16bf;
typedef __attribute__((ext_vector_type(8)))  float           v8f;
typedef __attribute__((ext_vector_type(8)))  unsigned short  v8us;
typedef __attribute__((ext_vector_type(4)))  unsigned int    u32x4;
typedef __attribute__((ext_vector_type(8)))  int             i32x8;
typedef __attribute__((ext_vector_type(4)))  int             i32x4;

union FragU {
    v8us  h[2];   // two 16-byte halves
    v16bf v;      // 32-byte WMMA operand
};

#define GCN_B   128
#define GCN_N   1024
#define GCN_F   128
#define GCN_D   256
#define GCN_M   (GCN_B * GCN_N)          // 131072 total nodes
#define EPSV    1e-5f

#if __has_builtin(__builtin_amdgcn_tensor_load_to_lds)
#define HAVE_TDM 1
#else
#define HAVE_TDM 0
#endif

// ---------------------------------------------------------------------------
// fp32 -> bf16 (round to nearest even)
// ---------------------------------------------------------------------------
__device__ __forceinline__ unsigned short f32_to_bf16(float f) {
    unsigned int u = __float_as_uint(f);
    unsigned int r = u + 0x7FFFu + ((u >> 16) & 1u);
    return (unsigned short)(r >> 16);
}

__global__ __launch_bounds__(256)
void k_convert_bf16(const float* __restrict__ src, unsigned short* __restrict__ dst, int n) {
    int i = blockIdx.x * blockDim.x + threadIdx.x;
    if (i < n) dst[i] = f32_to_bf16(src[i]);
}

// W [K x Nw] fp32 -> Wt [Nw x K] bf16 (transposed so GEMM B-tiles are
// contiguous row-major 2D tiles for the TDM)
__global__ __launch_bounds__(256)
void k_convert_transpose_bf16(const float* __restrict__ src, unsigned short* __restrict__ dst,
                              int K, int Nw) {
    int i = blockIdx.x * blockDim.x + threadIdx.x;
    if (i < K * Nw) {
        int k = i / Nw, n = i % Nw;
        dst[(size_t)n * K + k] = f32_to_bf16(src[i]);
    }
}

// ---------------------------------------------------------------------------
// Degree / dinv over the N base nodes (edge list is replicated per graph)
// ---------------------------------------------------------------------------
__global__ void k_deg_init(float* degN, int n) {
    int i = blockIdx.x * blockDim.x + threadIdx.x;
    if (i < n) degN[i] = 1.0f;   // self loop
}

__global__ void k_deg_count(const int* __restrict__ ei, int E, float* degN) {
    int e = blockIdx.x * blockDim.x + threadIdx.x;
    if (e < E) atomicAdd(&degN[ei[E + e]], 1.0f);   // row 1 = dst
}

__global__ void k_dinv(const float* __restrict__ degN, float* __restrict__ dinvN, int n) {
    int i = blockIdx.x * blockDim.x + threadIdx.x;
    if (i < n) dinvN[i] = rsqrtf(degN[i]);
}

// ---------------------------------------------------------------------------
// Tensor Data Mover: issue a 2D-tile load into LDS (D# per ISA 8.3-8.5)
//   data_size = 2 bytes (bf16); tile of tile1 rows x tile0 elems,
//   row stride in the global tensor = stride0 elems.
//   6-arg builtin (clang-23 lane): (g0, g1, g2, g3, g_extra, cpol)
// ---------------------------------------------------------------------------
__device__ __forceinline__ void tdm_load_tile_2d(unsigned lds_off, const void* gaddr,
                                                 unsigned dim0, unsigned dim1,
                                                 unsigned tile0, unsigned tile1,
                                                 unsigned stride0) {
#if HAVE_TDM
    unsigned long long ga = (unsigned long long)(size_t)gaddr;
    u32x4 g0;
    g0[0] = 1u;                                                  // count=1, user D#
    g0[1] = lds_off;                                             // lds_addr (bytes)
    g0[2] = (unsigned)(ga & 0xFFFFFFFFull);                      // global_addr[31:0]
    g0[3] = (unsigned)((ga >> 32) & 0x01FFFFFFull) | (2u << 30); // [56:32] | type=2
    i32x8 g1;
    g1[0] = (int)(1u << 16);                                     // data_size=2B
    g1[1] = (int)((dim0 & 0xFFFFu) << 16);                       // tensor_dim0[15:0]
    g1[2] = (int)(((dim0 >> 16) & 0xFFFFu) | ((dim1 & 0xFFFFu) << 16)); // dim0 hi | dim1 lo
    g1[3] = (int)(((dim1 >> 16) & 0xFFFFu) | ((tile0 & 0xFFFFu) << 16)); // dim1 hi | tile_dim0
    g1[4] = (int)(tile1 & 0xFFFFu);                              // tile_dim1 (tile_dim2=0)
    g1[5] = (int)stride0;                                        // tensor_dim0_stride[31:0]
    g1[6] = 0;
    g1[7] = 0;
    i32x4 z4 = {0, 0, 0, 0};
    i32x8 z8 = {0, 0, 0, 0, 0, 0, 0, 0};
    __builtin_amdgcn_tensor_load_to_lds(g0, g1, z4, z4, z8, 0);
#else
    (void)lds_off; (void)gaddr; (void)dim0; (void)dim1;
    (void)tile0; (void)tile1; (void)stride0;
#endif
}

// ---------------------------------------------------------------------------
// WMMA bf16 GEMM:  C[M x Nw] = A[M x K] * Bt[Nw x K]^T   (A, Bt row-major)
// Block tile 128x64, 8 waves (4x2), each wave computes 32x32 via 2x2 WMMA
// 16x16x32 bf16 tiles, fp32 accumulate. Tiles staged in LDS by the TDM.
// ---------------------------------------------------------------------------
#define BM 128
#define BN 64
#define BK 32

__global__ __launch_bounds__(256)
void k_gemm_bf16_wmma(const unsigned short* __restrict__ A,
                      const unsigned short* __restrict__ Bt,
                      float* __restrict__ C,
                      int M, int K, int Nw)
{
    __shared__ unsigned short sA [BM][BK];   // 128x32 bf16 = 8 KB
    __shared__ unsigned short sBt[BN][BK];   // 64x32 bf16 = 4 KB ([n][k])

    const int tid  = threadIdx.x;
    const int lane = tid & 31;
    const int wave = tid >> 5;      // 0..7
    const int wm   = wave & 3;      // 0..3 : M offset 32*wm
    const int wn   = wave >> 2;     // 0..1 : N offset 32*wn

    const int bm = blockIdx.x * BM;
    const int bn = blockIdx.y * BN;

    const unsigned sA_off  = (unsigned)(size_t)(void*)&sA[0][0];
    const unsigned sBt_off = (unsigned)(size_t)(void*)&sBt[0][0];

    v8f acc[2][2] = {};

    const int halfsel = (lane < 16) ? 0 : 1;     // wave32 half
    const int lr      = lane & 15;

    for (int k0 = 0; k0 < K; k0 += BK) {
#if HAVE_TDM
        // ---- TDM stages both tiles; wave 0 issues, everyone syncs ---------
        if (wave == 0) {
            tdm_load_tile_2d(sA_off,  A  + (size_t)bm * K + k0, K, M,  BK, BM, K);
            tdm_load_tile_2d(sBt_off, Bt + (size_t)bn * K + k0, K, Nw, BK, BN, K);
            __builtin_amdgcn_s_wait_tensorcnt((short)0);
        } else if (wave == 7 && k0 + BK < K) {
            // prefetch a slice of the next A tile (global_prefetch_b8)
            __builtin_prefetch(A + (size_t)(bm + lane * 4) * K + k0 + BK, 0, 0);
        }
#else
        // ---- fallback: manual staging through VGPRs -----------------------
        {
            int r  = tid >> 1;               // 0..127
            int cs = (tid & 1) * 16;         // 0 or 16
            const unsigned short* src = A + (size_t)(bm + r) * K + k0 + cs;
            *(v8us*)&sA[r][cs]     = *(const v8us*)(src);
            *(v8us*)&sA[r][cs + 8] = *(const v8us*)(src + 8);
        }
        {
            int n  = tid >> 2;               // 0..63
            int ks = (tid & 3) * 8;          // 0..24
            *(v8us*)&sBt[n][ks] = *(const v8us*)(Bt + (size_t)(bn + n) * K + k0 + ks);
        }
#endif
        __syncthreads();

        // ---- build fragments in contiguous k order (k = 16*half + s) -----------------------
        // A 16x32 bf16: lanes 0-15 hold K 0-15; lanes 16-31 hold
        //               K 16-31 (row M = lane%16).
        FragU afrag[2], bfrag[2];
        const int ka = halfsel * 16;         // 0 or 16
#pragma unroll
        for (int mi = 0; mi < 2; ++mi) {
            int r = wm * 32 + mi * 16 + lr;
            afrag[mi].h[0] = *(const v8us*)&sA[r][ka];
            afrag[mi].h[1] = *(const v8us*)&sA[r][ka + 8];
        }
        // B 32x16 bf16: lanes 0-15 hold K 0-15, lanes 16-31 hold K 16-31
        //               (col N = lane%16) -> contiguous in transposed tile.
        const int kb = halfsel * 16;         // 0 or 16
#pragma unroll
        for (int ni = 0; ni < 2; ++ni) {
            int n = wn * 32 + ni * 16 + lr;
            bfrag[ni].h[0] = *(const v8us*)&sBt[n][kb];
            bfrag[ni].h[1] = *(const v8us*)&sBt[n][kb + 8];
        }

#pragma unroll
        for (int mi = 0; mi < 2; ++mi)
#pragma unroll
            for (int ni = 0; ni < 2; ++ni)
                acc[mi][ni] = __builtin_amdgcn_wmma_f32_16x16x32_bf16(
                    /*neg_a=*/false, afrag[mi].v,
                    /*neg_b=*/false, bfrag[ni].v,
                    /*c_mod=*/(short)0, acc[mi][ni],
                    /*reuse_a=*/false, /*reuse_b=*/false);
        __syncthreads();
    }

    // ---- store C (16x16 f32 layout: VGPR v -> M = v + 8*halfsel) ----------
#pragma unroll
    for (int mi = 0; mi < 2; ++mi)
#pragma unroll
        for (int ni = 0; ni < 2; ++ni) {
            int row = bm + wm * 32 + mi * 16 + halfsel * 8;
            int col = bn + wn * 32 + ni * 16 + lr;
            float* cp = C + (size_t)row * Nw + col;
#pragma unroll
            for (int v = 0; v < 8; ++v)
                cp[(size_t)v * Nw] = acc[mi][ni][v];
        }
}

// ---------------------------------------------------------------------------
// Aggregation: agg = self-loop term; then edge scatter-add
// ---------------------------------------------------------------------------
__global__ __launch_bounds__(256)
void k_self_init(const float* __restrict__ hw, const float* __restrict__ dinvN,
                 float* __restrict__ agg, int total /* M*D */) {
    int i = blockIdx.x * blockDim.x + threadIdx.x;
    if (i < total) {
        int node = i / GCN_D;
        float di = dinvN[node & (GCN_N - 1)];
        agg[i] = hw[i] * di * di;
    }
}

__global__ __launch_bounds__(256)
void k_edge_agg(const int* __restrict__ ei, const float* __restrict__ dinvN,
                const float* __restrict__ hw, float* __restrict__ agg, int E) {
    long long t = (long long)blockIdx.x * blockDim.x + threadIdx.x;
    long long totalw = (long long)GCN_B * E * GCN_D;
    if (t >= totalw) return;
    int d    = (int)(t % GCN_D);
    long long r = t / GCN_D;
    int e    = (int)(r % E);
    int b    = (int)(r / E);
    int src  = ei[e];
    int dst  = ei[E + e];
    float w  = dinvN[src] * dinvN[dst];
    size_t si = ((size_t)b * GCN_N + src) * GCN_D + d;
    size_t di = ((size_t)b * GCN_N + dst) * GCN_D + d;
    atomicAdd(&agg[di], hw[si] * w);
}

// ---------------------------------------------------------------------------
// BatchNorm statistics + apply
// ---------------------------------------------------------------------------
__global__ void k_zero_stats(float* sums, float* sumsq) {
    int d = threadIdx.x;
    if (d < GCN_D) { sums[d] = 0.0f; sumsq[d] = 0.0f; }
}

__global__ __launch_bounds__(256)
void k_bias_relu_stats(const float* __restrict__ agg, const float* __restrict__ bias,
                       float* __restrict__ t, float* __restrict__ sums,
                       float* __restrict__ sumsq, int total) {
    int i = blockIdx.x * blockDim.x + threadIdx.x;
    if (i < total) {
        int d = i & (GCN_D - 1);
        float v = agg[i] + bias[d];
        v = fmaxf(v, 0.0f);
        t[i] = v;
        atomicAdd(&sums[d], v);
        atomicAdd(&sumsq[d], v * v);
    }
}

__global__ __launch_bounds__(256)
void k_bn_apply_bf16(const float* __restrict__ t, const float* __restrict__ sums,
                     const float* __restrict__ sumsq, const float* __restrict__ g,
                     const float* __restrict__ bt, unsigned short* __restrict__ out,
                     int total, float invM) {
    int i = blockIdx.x * blockDim.x + threadIdx.x;
    if (i < total) {
        int d = i & (GCN_D - 1);
        float mu  = sums[d] * invM;
        float var = sumsq[d] * invM - mu * mu;
        float y = (t[i] - mu) * rsqrtf(var + EPSV) * g[d] + bt[d];
        out[i] = f32_to_bf16(y);
    }
}

__global__ __launch_bounds__(256)
void k_bias_relu_out(const float* __restrict__ agg, const float* __restrict__ bias,
                     float* __restrict__ out, int total) {
    int i = blockIdx.x * blockDim.x + threadIdx.x;
    if (i < total) {
        int d = i & (GCN_D - 1);
        out[i] = fmaxf(agg[i] + bias[d], 0.0f);
    }
}

// ---------------------------------------------------------------------------
// Host-side orchestration
// ---------------------------------------------------------------------------
static inline size_t align256(size_t x) { return (x + 255) & ~(size_t)255; }

extern "C" void kernel_launch(void* const* d_in, const int* in_sizes, int n_in,
                              void* d_out, int out_size, void* d_ws, size_t ws_size,
                              hipStream_t stream) {
    (void)n_in; (void)out_size; (void)ws_size;

    const float* x   = (const float*)d_in[0];     // [B,N,F]
    const int*   ei  = (const int*)d_in[1];       // [2,E]
    const float* W1  = (const float*)d_in[3];
    const float* b1  = (const float*)d_in[4];
    const float* W2  = (const float*)d_in[5];
    const float* b2  = (const float*)d_in[6];
    const float* W3  = (const float*)d_in[7];
    const float* b3  = (const float*)d_in[8];
    const float* g1  = (const float*)d_in[9];
    const float* bt1 = (const float*)d_in[10];
    const float* g2  = (const float*)d_in[11];
    const float* bt2 = (const float*)d_in[12];

    const int E = in_sizes[1] / 2;                // 8192
    const int M = GCN_M;                          // 131072
    const int D = GCN_D;                          // 256
    const int F = GCN_F;                          // 128
    const size_t MD = (size_t)M * D;              // 33.5M elems

    // ---- workspace carve-out ----------------------------------------------
    char* p = (char*)d_ws;
    float*          hw    = (float*)p;            p += align256(MD * 4);   // gemm out / relu buf
    float*          agg   = (float*)p;            p += align256(MD * 4);
    unsigned short* inbf  = (unsigned short*)p;   p += align256(MD * 2);   // bf16 activations
    unsigned short* w1t   = (unsigned short*)p;   p += align256((size_t)F * D * 2); // [D x F]
    unsigned short* w2t   = (unsigned short*)p;   p += align256((size_t)D * D * 2); // [D x D]
    unsigned short* w3t   = (unsigned short*)p;   p += align256((size_t)D * D * 2);
    float*          degN  = (float*)p;            p += align256(GCN_N * 4);
    float*          dinvN = (float*)p;            p += align256(GCN_N * 4);
    float*          sums  = (float*)p;            p += align256(D * 4);
    float*          sumsq = (float*)p;            p += align256(D * 4);

    const int T = 256;
    const float invM = 1.0f / (float)M;

    // ---- precision conversion (fp32 -> bf16, weights transposed) ----------
    {
        int n = M * F;
        k_convert_bf16<<<(n + T - 1) / T, T, 0, stream>>>(x, inbf, n);
        n = F * D;
        k_convert_transpose_bf16<<<(n + T - 1) / T, T, 0, stream>>>(W1, w1t, F, D);
        n = D * D;
        k_convert_transpose_bf16<<<(n + T - 1) / T, T, 0, stream>>>(W2, w2t, D, D);
        k_convert_transpose_bf16<<<(n + T - 1) / T, T, 0, stream>>>(W3, w3t, D, D);
    }

    // ---- degrees / normalization ------------------------------------------
    k_deg_init<<<(GCN_N + T - 1) / T, T, 0, stream>>>(degN, GCN_N);
    k_deg_count<<<(E + T - 1) / T, T, 0, stream>>>(ei, E, degN);
    k_dinv<<<(GCN_N + T - 1) / T, T, 0, stream>>>(degN, dinvN, GCN_N);

    const long long edgeWork = (long long)GCN_B * E * D;
    const int edgeBlocks = (int)((edgeWork + T - 1) / T);
    const int mdBlocks   = (int)((MD + T - 1) / T);

    dim3 gemmBlocks(M / BM, D / BN);  // (1024, 4)

    // ======================= Layer 1 =======================================
    k_gemm_bf16_wmma<<<gemmBlocks, 256, 0, stream>>>(inbf, w1t, hw, M, F, D);
    k_self_init<<<mdBlocks, T, 0, stream>>>(hw, dinvN, agg, (int)MD);
    k_edge_agg<<<edgeBlocks, T, 0, stream>>>(ei, dinvN, hw, agg, E);
    k_zero_stats<<<1, 256, 0, stream>>>(sums, sumsq);
    k_bias_relu_stats<<<mdBlocks, T, 0, stream>>>(agg, b1, hw, sums, sumsq, (int)MD);
    k_bn_apply_bf16<<<mdBlocks, T, 0, stream>>>(hw, sums, sumsq, g1, bt1, inbf, (int)MD, invM);

    // ======================= Layer 2 =======================================
    k_gemm_bf16_wmma<<<gemmBlocks, 256, 0, stream>>>(inbf, w2t, hw, M, D, D);
    k_self_init<<<mdBlocks, T, 0, stream>>>(hw, dinvN, agg, (int)MD);
    k_edge_agg<<<edgeBlocks, T, 0, stream>>>(ei, dinvN, hw, agg, E);
    k_zero_stats<<<1, 256, 0, stream>>>(sums, sumsq);
    k_bias_relu_stats<<<mdBlocks, T, 0, stream>>>(agg, b2, hw, sums, sumsq, (int)MD);
    k_bn_apply_bf16<<<mdBlocks, T, 0, stream>>>(hw, sums, sumsq, g2, bt2, inbf, (int)MD, invM);

    // ======================= Layer 3 =======================================
    k_gemm_bf16_wmma<<<gemmBlocks, 256, 0, stream>>>(inbf, w3t, hw, M, D, D);
    k_self_init<<<mdBlocks, T, 0, stream>>>(hw, dinvN, agg, (int)MD);
    k_edge_agg<<<edgeBlocks, T, 0, stream>>>(ei, dinvN, hw, agg, E);
    k_bias_relu_out<<<mdBlocks, T, 0, stream>>>(agg, b3, (float*)d_out, (int)MD);
}
